// SentenceQwen2DecoderLayer_16372415332650
// MI455X (gfx1250) — compile-verified, matrix operand order checked by reading, not run on hardware
//
#include <hip/hip_runtime.h>
#include <hip/hip_bf16.h>

// ---------------------------------------------------------------------------
// Qwen2 decoder layer for MI455X (gfx1250, wave32, WMMA + TDM).
// Memory-bound on fp32 weight streaming (~320MB @ 23.3TB/s ~= 15us), so:
//  - every weight is read exactly once (fp32 -> bf16 convert into LDS)
//  - all matmuls go through v_wmma_f32_16x16x32_bf16 (f32 accumulate)
//  - bf16 activation tiles are DMA'd global->LDS by the Tensor Data Mover
//    (tensor_load_to_lds, TENSORcnt), freeing vector issue slots
//  - flash-attention blocks avoid materializing the 1GB score tensor
// ---------------------------------------------------------------------------

typedef __bf16 bf16;
typedef __attribute__((ext_vector_type(16))) __bf16 v16bf;
typedef __attribute__((ext_vector_type(8)))  float  v8f;
typedef __attribute__((ext_vector_type(4)))  unsigned int u32x4;
typedef __attribute__((ext_vector_type(8)))  unsigned int u32x8;

#define B_  2
#define S_  2048
#define HS_ 2048
#define H_  16
#define KV_ 8
#define D_  128
#define I_  8192
#define M_  (B_ * S_)   // 4096 token rows

union FragU { v16bf v; uint4 q[2]; };

__device__ inline v8f v8f_zero() {
  v8f z;
#pragma unroll
  for (int i = 0; i < 8; ++i) z[i] = 0.0f;
  return z;
}

// Load a 16x32 bf16 fragment (A-operand layout, ISA 7.12.2) from a row-major
// LDS tile. For the B-operand we store the tile N-major (i.e. W^T), so the
// same loader produces the 32x16 B layout. lda must be a multiple of 8.
__device__ inline v16bf load_frag(const bf16* __restrict__ base, int lda) {
  const int lane = threadIdx.x & 31;
  const int r    = lane & 15;   // M (or N) row
  const int half = lane >> 4;   // K-half select
  const bf16* p = base + r * lda + half * 8;
  FragU u;
  u.q[0] = *(const uint4*)(p);       // K = half*8 .. half*8+7
  u.q[1] = *(const uint4*)(p + 16);  // K = 16+half*8 .. +7
  return u.v;
}

// ---------------------------------------------------------------------------
// Tensor Data Mover: DMA a 2-D bf16 tile (tile_d0 elems per row, tile_d1 rows,
// row stride stride0 elems) from global into LDS, inserting pad_amount DWORDs
// of padding after each 2^pad_interval-coded DWORD interval (ISA 8.3-8.4).
// Issued by one wave; EXEC is ignored by TDM; completion via TENSORcnt.
// ---------------------------------------------------------------------------
__device__ __attribute__((always_inline)) inline void
tdm_load_2d(unsigned lds_addr, const void* gaddr,
            unsigned tile_d0, unsigned tile_d1,
            unsigned tensor_d0, unsigned tensor_d1,
            unsigned long long stride0,
            unsigned pad_interval, unsigned pad_amount) {
  const unsigned long long ga = (unsigned long long)(uintptr_t)gaddr;
  u32x4 g0;
  g0[0] = 1u;                                        // count=1, user descriptor
  g0[1] = lds_addr;                                  // LDS byte address
  g0[2] = (unsigned)(ga & 0xffffffffu);              // global_addr[31:0]
  g0[3] = (unsigned)((ga >> 32) & 0x01ffffffu) |     // global_addr[56:32]
          (2u << 30);                                // type = 2 ("image")
  u32x8 g1;
  g1[0] = (1u << 16)                                 // data_size = 2 bytes
        | (1u << 20)                                 // pad_enable
        | (pad_interval << 22)
        | (pad_amount << 25);
  g1[1] = (tensor_d0 & 0xffffu) << 16;               // tensor_dim0[15:0]
  g1[2] = ((tensor_d0 >> 16) & 0xffffu)              // tensor_dim0[31:16]
        | ((tensor_d1 & 0xffffu) << 16);             // tensor_dim1[15:0]
  g1[3] = ((tensor_d1 >> 16) & 0xffffu)              // tensor_dim1[31:16]
        | (tile_d0 << 16);                           // tile_dim0
  g1[4] = tile_d1 & 0xffffu;                         // tile_dim1 (tile_dim2=0)
  g1[5] = (unsigned)(stride0 & 0xffffffffu);         // tensor_dim0_stride[31:0]
  g1[6] = (unsigned)((stride0 >> 32) & 0xffffu);     // stride[47:32]; dim1_stride=0
  g1[7] = 0u;
  asm volatile("tensor_load_to_lds %0, %1" :: "s"(g0), "s"(g1) : "memory");
}

__device__ inline unsigned lds_byte_addr(const void* p) {
  return (unsigned)(uintptr_t)p;
}

// ---------------------------------------------------------------------------
// RMSNorm: fp32 in -> bf16 out (feeds WMMA GEMMs). One block per token row.
// ---------------------------------------------------------------------------
__launch_bounds__(256)
__global__ void rmsnorm_kernel(const float* __restrict__ x,
                               const float* __restrict__ w,
                               bf16* __restrict__ out) {
  __shared__ float red[256];
  const size_t row = blockIdx.x;
  const float* xr = x + row * HS_;
  const int tid = threadIdx.x;
  float vals[8];
  float ss = 0.0f;
#pragma unroll
  for (int i = 0; i < 8; ++i) {
    float v = xr[tid + i * 256];
    vals[i] = v;
    ss += v * v;
  }
  red[tid] = ss;
  __syncthreads();
  for (int s2 = 128; s2 > 0; s2 >>= 1) {
    if (tid < s2) red[tid] += red[tid + s2];
    __syncthreads();
  }
  const float scale = rsqrtf(red[0] * (1.0f / HS_) + 1e-6f);
#pragma unroll
  for (int i = 0; i < 8; ++i) {
    const int c = tid + i * 256;
    out[row * HS_ + c] = (bf16)(vals[i] * scale * w[c]);
  }
}

// ---------------------------------------------------------------------------
// GEMM: C[M,N] = A_bf16[M,K] @ W_f32[K,N] (+bias) with fused epilogues.
//  EPI=0: fp32 out (+bias)
//  EPI=1: fp32 out = acc + aux (residual add)
//  EPI=2: bf16 out = silu(aux) * acc   (SwiGLU; aux = gate activations)
// Block tile 128x128x32, 8 waves (4x2), 8 WMMAs per wave per k-step.
// A tile arrives via TDM (row pad 16B -> lda 40); W tile is converted
// fp32->bf16 in registers and stored N-major so B-frags are contiguous.
// ---------------------------------------------------------------------------
template <int EPI>
__launch_bounds__(256)
__global__ void gemm_bf16_kernel(const bf16*  __restrict__ A,
                                 const float* __restrict__ W,
                                 const float* __restrict__ bias,
                                 const float* __restrict__ aux,
                                 float* __restrict__ Cf,
                                 bf16*  __restrict__ Cb,
                                 int M, int N, int K) {
  __shared__ __attribute__((aligned(16))) bf16 sA[128 * 40];  // [m][k], lda 40
  __shared__ __attribute__((aligned(16))) bf16 sB[128 * 40];  // [n][k] (W^T)

  const int tid = threadIdx.x;
  const int wid = tid >> 5;
  const int wm  = wid & 3;    // 4 row groups of 32
  const int wn  = wid >> 2;   // 2 col groups of 64
  const int m0  = blockIdx.y * 128;
  const int n0  = blockIdx.x * 128;
  const unsigned sA_lds = lds_byte_addr(&sA[0]);

  v8f acc[2][4];
#pragma unroll
  for (int i = 0; i < 2; ++i)
#pragma unroll
    for (int j = 0; j < 4; ++j) acc[i][j] = v8f_zero();

  const int ksteps = K >> 5;
  for (int kt = 0; kt < ksteps; ++kt) {
    const int k0 = kt << 5;
    // TDM: A tile 128 rows x 32 cols bf16, row stride K, 16B pad per 64B row
    if (wid == 0) {
      tdm_load_2d(sA_lds, A + (size_t)m0 * K + k0,
                  /*tile_d0=*/32, /*tile_d1=*/128,
                  /*tensor_d0=*/(unsigned)K, /*tensor_d1=*/(unsigned)M,
                  /*stride0=*/(unsigned long long)K,
                  /*pad_interval=*/3, /*pad_amount=*/3);
    }
    if (kt + 1 < ksteps) {
      // gfx1250 global_prefetch_b8 for the next weight tile
      __builtin_prefetch(W + (size_t)(k0 + 32 + (tid >> 3)) * N + n0, 0, 0);
    }
    // W tile: 32x128 fp32 -> bf16, stored transposed as sB[n][k]
#pragma unroll
    for (int i = 0; i < 4; ++i) {
      const int c  = tid + i * 256;       // 1024 float4 chunks
      const int kr = c >> 5;              // 0..31
      const int nn = (c & 31) << 2;       // 0..124
      const float4 wv4 = *(const float4*)(W + (size_t)(k0 + kr) * N + n0 + nn);
      sB[(nn + 0) * 40 + kr] = (bf16)wv4.x;
      sB[(nn + 1) * 40 + kr] = (bf16)wv4.y;
      sB[(nn + 2) * 40 + kr] = (bf16)wv4.z;
      sB[(nn + 3) * 40 + kr] = (bf16)wv4.w;
    }
    if (wid == 0) __builtin_amdgcn_s_wait_tensorcnt(0);  // A tile landed
    __syncthreads();

    v16bf afrag0 = load_frag(&sA[(wm * 32 + 0)  * 40], 40);
    v16bf afrag1 = load_frag(&sA[(wm * 32 + 16) * 40], 40);
#pragma unroll
    for (int tn = 0; tn < 4; ++tn) {
      v16bf bfrag = load_frag(&sB[(wn * 64 + tn * 16) * 40], 40);
      acc[0][tn] = __builtin_amdgcn_wmma_f32_16x16x32_bf16(
          false, afrag0, false, bfrag, (short)0, acc[0][tn], false, false);
      acc[1][tn] = __builtin_amdgcn_wmma_f32_16x16x32_bf16(
          false, afrag1, false, bfrag, (short)0, acc[1][tn], false, false);
    }
    __syncthreads();
  }

  // Epilogue. C/D layout (ISA 7.12.2): VGPR r -> M = (lane>>4)*8 + r, N = lane&15
  const int lane = tid & 31;
  const int mo = (lane >> 4) * 8;
  const int nn = lane & 15;
#pragma unroll
  for (int tm = 0; tm < 2; ++tm)
#pragma unroll
    for (int tn = 0; tn < 4; ++tn)
#pragma unroll
      for (int r = 0; r < 8; ++r) {
        const int m = m0 + wm * 32 + tm * 16 + mo + r;
        const int n = n0 + wn * 64 + tn * 16 + nn;
        float v = acc[tm][tn][r];
        if (bias) v += bias[n];
        const size_t idx = (size_t)m * N + n;
        if (EPI == 0) {
          Cf[idx] = v;
        } else if (EPI == 1) {
          Cf[idx] = v + aux[idx];
        } else {
          const float g  = aux[idx];
          const float sg = g / (1.0f + __expf(-g));  // silu(gate)
          Cb[idx] = (bf16)(sg * v);
        }
      }
}

// ---------------------------------------------------------------------------
// RoPE + head-major transpose: fp32 QKV projections -> bf16 [b,h,s,d].
// ---------------------------------------------------------------------------
__launch_bounds__(256)
__global__ void rope_kernel(const float* __restrict__ qf,
                            const float* __restrict__ kf,
                            const float* __restrict__ vf,
                            const float* __restrict__ cosb,
                            const float* __restrict__ sinb,
                            bf16* __restrict__ qr,
                            bf16* __restrict__ kr,
                            bf16* __restrict__ vr) {
  const int bs = blockIdx.x;           // b*S + s
  const int b = bs / S_, s = bs % S_;
  const float* cs = cosb + (size_t)bs * D_;
  const float* sn = sinb + (size_t)bs * D_;
  const int tid = threadIdx.x;
  for (int i = tid; i < H_ * D_; i += 256) {
    const int h = i >> 7, d = i & 127;
    const float* row = qf + (size_t)bs * (H_ * D_) + h * D_;
    const float x = row[d];
    const float o = (d < 64) ? -row[d + 64] : row[d - 64];
    qr[((size_t)(b * H_ + h) * S_ + s) * D_ + d] = (bf16)(x * cs[d] + o * sn[d]);
  }
  for (int i = tid; i < KV_ * D_; i += 256) {
    const int h = i >> 7, d = i & 127;
    const float* row = kf + (size_t)bs * (KV_ * D_) + h * D_;
    const float x = row[d];
    const float o = (d < 64) ? -row[d + 64] : row[d - 64];
    const size_t dst = ((size_t)(b * KV_ + h) * S_ + s) * D_ + d;
    kr[dst] = (bf16)(x * cs[d] + o * sn[d]);
    vr[dst] = (bf16)vf[(size_t)bs * (KV_ * D_) + i];
  }
}

// ---------------------------------------------------------------------------
// Flash attention, 32-query x 64-key blocks, WMMA for QK^T and P*V, online
// softmax with causal + (special | k>=eos | k==0) masking. GQA group = 2.
// Q/K tiles DMA'd by TDM (row pad 16B -> lda 136); V transposed manually.
// Static LDS ~56KB.
// ---------------------------------------------------------------------------
__launch_bounds__(256)
__global__ void flash_attn_kernel(const bf16* __restrict__ Q,
                                  const bf16* __restrict__ K,
                                  const bf16* __restrict__ V,
                                  const unsigned char* __restrict__ special,
                                  const int* __restrict__ eos,
                                  bf16* __restrict__ O) {
  __shared__ __attribute__((aligned(16))) bf16  sQ[32 * 136];   // [q][d]
  __shared__ __attribute__((aligned(16))) bf16  sK[64 * 136];   // [k][d]
  __shared__ __attribute__((aligned(16))) bf16  sVt[128 * 72];  // [d][k] (V^T)
  __shared__ __attribute__((aligned(16))) float sS[32 * 64];    // scores
  __shared__ __attribute__((aligned(16))) bf16  sP[32 * 72];    // probs
  __shared__ float m_s[32], l_s[32], alpha_s[32];

  const int tid = threadIdx.x;
  const int wid = tid >> 5, lane = tid & 31;
  const int qb = blockIdx.x;                 // S/32 query blocks
  const int bh = blockIdx.y;                 // B*H
  const int b = bh / H_, h = bh % H_;
  const int kvh = h / (H_ / KV_);
  const float scaling = 0.08838834764831845f;  // 1/sqrt(128)

  const bf16* Qb = Q + (size_t)(b * H_  + h)   * S_ * D_;
  const bf16* Kb = K + (size_t)(b * KV_ + kvh) * S_ * D_;
  const bf16* Vb = V + (size_t)(b * KV_ + kvh) * S_ * D_;
  const int q0 = qb * 32;

  // TDM: Q tile 32x128 (waited together with the first K tile)
  if (wid == 0) {
    tdm_load_2d(lds_byte_addr(&sQ[0]), Qb + (size_t)q0 * D_,
                /*tile_d0=*/128, /*tile_d1=*/32,
                /*tensor_d0=*/D_, /*tensor_d1=*/S_,
                /*stride0=*/(unsigned long long)D_,
                /*pad_interval=*/5, /*pad_amount=*/3);
  }
  if (tid < 32) { m_s[tid] = -1e30f; l_s[tid] = 0.0f; }

  // O accumulators: wave covers 16 q rows x 32 d cols
  const int wqm = (wid & 1) * 16;
  const int wd  = (wid >> 1) * 32;
  v8f oacc[2];
  oacc[0] = v8f_zero();
  oacc[1] = v8f_zero();

  const int nkb = (q0 + 31) / 64 + 1;
  for (int kb = 0; kb < nkb; ++kb) {
    const int k0 = kb * 64;
    __syncthreads();  // prev P*V done before K/V tiles are overwritten

    // TDM: K tile 64x128
    if (wid == 0) {
      tdm_load_2d(lds_byte_addr(&sK[0]), Kb + (size_t)k0 * D_,
                  /*tile_d0=*/128, /*tile_d1=*/64,
                  /*tensor_d0=*/D_, /*tensor_d1=*/S_,
                  /*stride0=*/(unsigned long long)D_,
                  /*pad_interval=*/5, /*pad_amount=*/3);
    }
    // V tile 64x128 -> LDS transposed [d][k] so P*V B-frags are contiguous
#pragma unroll
    for (int i = 0; i < 4; ++i) {
      const int c = tid + i * 256;
      const int row = c >> 4;
      const int dc = (c & 15) << 3;
      const uint4 dv = *(const uint4*)(Vb + (size_t)(k0 + row) * D_ + dc);
      const bf16* e = (const bf16*)&dv;
#pragma unroll
      for (int j = 0; j < 8; ++j) sVt[(dc + j) * 72 + row] = e[j];
    }
    if (wid == 0) __builtin_amdgcn_s_wait_tensorcnt(0);  // Q (iter 0) + K landed
    __syncthreads();

    // S = Q K^T : each wave computes one 16x16 tile
    {
      const int sqm = (wid & 1) * 16;
      const int skn = (wid >> 1) * 16;
      v8f sacc = v8f_zero();
#pragma unroll
      for (int dt = 0; dt < 4; ++dt) {
        v16bf aq = load_frag(&sQ[sqm * 136 + dt * 32], 136);
        v16bf bk = load_frag(&sK[skn * 136 + dt * 32], 136);
        sacc = __builtin_amdgcn_wmma_f32_16x16x32_bf16(
            false, aq, false, bk, (short)0, sacc, false, false);
      }
      const int mo = (lane >> 4) * 8, nc = lane & 15;
#pragma unroll
      for (int r = 0; r < 8; ++r)
        sS[(sqm + mo + r) * 64 + skn + nc] = sacc[r];
    }
    __syncthreads();

    // Online softmax: one thread per query row
    if (tid < 32) {
      const int q = q0 + tid;
      const int eosq = eos[b * S_ + q];
      const float mold = m_s[tid];
      float mt = mold;
      for (int j = 0; j < 64; ++j) {
        const int kk = k0 + j;
        const bool allowed =
            (kk <= q) && (special[b * S_ + kk] != 0 || kk >= eosq || kk == 0);
        const float s = allowed ? sS[tid * 64 + j] * scaling : -1e9f;
        mt = fmaxf(mt, s);
      }
      const float alpha = __expf(mold - mt);
      float lsum = l_s[tid] * alpha;
      for (int j = 0; j < 64; ++j) {
        const int kk = k0 + j;
        const bool allowed =
            (kk <= q) && (special[b * S_ + kk] != 0 || kk >= eosq || kk == 0);
        const float s = allowed ? sS[tid * 64 + j] * scaling : -1e9f;
        const float pj = __expf(s - mt);
        lsum += pj;
        sP[tid * 72 + j] = (bf16)pj;
      }
      m_s[tid] = mt; l_s[tid] = lsum; alpha_s[tid] = alpha;
    }
    __syncthreads();

    // Rescale O and accumulate P*V
    {
      const int mo = (lane >> 4) * 8;
#pragma unroll
      for (int tn = 0; tn < 2; ++tn)
#pragma unroll
        for (int r = 0; r < 8; ++r)
          oacc[tn][r] *= alpha_s[wqm + mo + r];
#pragma unroll
      for (int ks = 0; ks < 2; ++ks) {
        v16bf ap = load_frag(&sP[wqm * 72 + ks * 32], 72);
#pragma unroll
        for (int tn = 0; tn < 2; ++tn) {
          v16bf bv = load_frag(&sVt[(wd + tn * 16) * 72 + ks * 32], 72);
          oacc[tn] = __builtin_amdgcn_wmma_f32_16x16x32_bf16(
              false, ap, false, bv, (short)0, oacc[tn], false, false);
        }
      }
    }
  }

  // Epilogue: O / l -> bf16 [m, h*D+d] for the O-projection GEMM
  const int mo = (lane >> 4) * 8, nc = lane & 15;
#pragma unroll
  for (int tn = 0; tn < 2; ++tn)
#pragma unroll
    for (int r = 0; r < 8; ++r) {
      const int m = wqm + mo + r;
      const int q = q0 + m;
      const int d = wd + tn * 16 + nc;
      const float val = oacc[tn][r] / l_s[m];
      O[(size_t)(b * S_ + q) * (H_ * D_) + h * D_ + d] = (bf16)val;
    }
}

// ---------------------------------------------------------------------------
// Launcher
// ---------------------------------------------------------------------------
extern "C" void kernel_launch(void* const* d_in, const int* in_sizes, int n_in,
                              void* d_out, int out_size, void* d_ws, size_t ws_size,
                              hipStream_t stream) {
  const float* hidden  = (const float*)d_in[0];
  const float* cosb    = (const float*)d_in[1];
  const float* sinb    = (const float*)d_in[2];
  const unsigned char* special = (const unsigned char*)d_in[3];
  const int*   eos     = (const int*)d_in[4];
  const float* ln1     = (const float*)d_in[5];
  const float* ln2     = (const float*)d_in[6];
  const float* wq      = (const float*)d_in[7];
  const float* bq      = (const float*)d_in[8];
  const float* wk      = (const float*)d_in[9];
  const float* bk      = (const float*)d_in[10];
  const float* wv      = (const float*)d_in[11];
  const float* bv      = (const float*)d_in[12];
  const float* wo      = (const float*)d_in[13];
  const float* wgate   = (const float*)d_in[14];
  const float* wup     = (const float*)d_in[15];
  const float* wdown   = (const float*)d_in[16];
  float* out = (float*)d_out;

  char* p = (char*)d_ws;
  auto alloc = [&](size_t bytes) {
    char* r = p;
    p += (bytes + 255) & ~(size_t)255;
    return r;
  };
  bf16*  hn    = (bf16*) alloc((size_t)M_ * HS_ * 2);
  float* qf    = (float*)alloc((size_t)M_ * (H_ * D_) * 4);
  float* kf    = (float*)alloc((size_t)M_ * (KV_ * D_) * 4);
  float* vf    = (float*)alloc((size_t)M_ * (KV_ * D_) * 4);
  bf16*  qr    = (bf16*) alloc((size_t)B_ * H_  * S_ * D_ * 2);
  bf16*  kr    = (bf16*) alloc((size_t)B_ * KV_ * S_ * D_ * 2);
  bf16*  vr    = (bf16*) alloc((size_t)B_ * KV_ * S_ * D_ * 2);
  bf16*  attn  = (bf16*) alloc((size_t)M_ * (H_ * D_) * 2);
  float* x1    = (float*)alloc((size_t)M_ * HS_ * 4);
  bf16*  hn2   = (bf16*) alloc((size_t)M_ * HS_ * 2);
  float* gate  = (float*)alloc((size_t)M_ * I_ * 4);
  bf16*  mlpin = (bf16*) alloc((size_t)M_ * I_ * 2);

  const dim3 blk(256);

  // 1) pre-attention RMSNorm -> bf16
  rmsnorm_kernel<<<M_, blk, 0, stream>>>(hidden, ln1, hn);
  // 2) QKV projections (WMMA, A tiles via TDM)
  gemm_bf16_kernel<0><<<dim3((H_ * D_) / 128, M_ / 128), blk, 0, stream>>>(
      hn, wq, bq, nullptr, qf, nullptr, M_, H_ * D_, HS_);
  gemm_bf16_kernel<0><<<dim3((KV_ * D_) / 128, M_ / 128), blk, 0, stream>>>(
      hn, wk, bk, nullptr, kf, nullptr, M_, KV_ * D_, HS_);
  gemm_bf16_kernel<0><<<dim3((KV_ * D_) / 128, M_ / 128), blk, 0, stream>>>(
      hn, wv, bv, nullptr, vf, nullptr, M_, KV_ * D_, HS_);
  // 3) RoPE + head-major bf16 transpose
  rope_kernel<<<B_ * S_, blk, 0, stream>>>(qf, kf, vf, cosb, sinb, qr, kr, vr);
  // 4) flash attention (WMMA, Q/K tiles via TDM)
  flash_attn_kernel<<<dim3(S_ / 32, B_ * H_), blk, 0, stream>>>(
      qr, kr, vr, special, eos, attn);
  // 5) O-projection + residual (WMMA, fused add)
  gemm_bf16_kernel<1><<<dim3(HS_ / 128, M_ / 128), blk, 0, stream>>>(
      attn, wo, nullptr, hidden, x1, nullptr, M_, HS_, H_ * D_);
  // 6) post-attention RMSNorm
  rmsnorm_kernel<<<M_, blk, 0, stream>>>(x1, ln2, hn2);
  // 7) MLP: gate, then up with fused SwiGLU -> bf16, then down + residual
  gemm_bf16_kernel<0><<<dim3(I_ / 128, M_ / 128), blk, 0, stream>>>(
      hn2, wgate, nullptr, nullptr, gate, nullptr, M_, I_, HS_);
  gemm_bf16_kernel<2><<<dim3(I_ / 128, M_ / 128), blk, 0, stream>>>(
      hn2, wup, nullptr, gate, nullptr, mlpin, M_, I_, HS_);
  gemm_bf16_kernel<1><<<dim3(HS_ / 128, M_ / 128), blk, 0, stream>>>(
      mlpin, wdown, nullptr, x1, out, nullptr, M_, HS_, I_);
}
